// Hydra_attn_16131897164212
// MI455X (gfx1250) — matrix operand order repaired and checked by reading, not compiled, not run on hardware
//
#include <hip/hip_runtime.h>
#include <hip/hip_bf16.h>

// ---------------------------------------------------------------------------
// MI455X (gfx1250) windowed-attention pipeline, f16 WMMA path.
//   x(8,64,64,512) -> qkv GEMM (wmma f16) -> RoPE+window attn (wmma f16)
//   -> proj GEMM (wmma f16) -> out fp32
// The reference's hy_kv branch (kv_w/kv_b) is dead code and is skipped.
// GEMMs use LDS double-buffering; tile staging goes through
// GLOBAL_LOAD_ASYNC_TO_LDS_B128 (ASYNCcnt) when the builtin exists, else
// global_load_b128 + ds_store_b128.
// ---------------------------------------------------------------------------

typedef __attribute__((ext_vector_type(16))) _Float16 v16h;
typedef __attribute__((ext_vector_type(8)))  _Float16 v8h;
typedef __attribute__((ext_vector_type(4)))  _Float16 v4h;
typedef __attribute__((ext_vector_type(8)))  float    v8f;
typedef __attribute__((ext_vector_type(4)))  float    v4f;
typedef __attribute__((ext_vector_type(4)))  int      v4i;

union V16 { v16h v; v8h h[2]; };

#if defined(__has_builtin)
#if __has_builtin(__builtin_amdgcn_global_load_async_to_lds_b128)
#define HAVE_ASYNC_LDS 1
#endif
#endif
#ifndef HAVE_ASYNC_LDS
#define HAVE_ASYNC_LDS 0
#endif

#if HAVE_ASYNC_LDS
// Probed signature: (int4* global_src, int4* lds_dst, imm offset, imm cpol)
#define ASYNC_COPY16(gp, lp)                                                   \
    __builtin_amdgcn_global_load_async_to_lds_b128((v4i*)(gp), (v4i*)(lp), 0, 0)
#if __has_builtin(__builtin_amdgcn_s_wait_asynccnt)
#define WAIT_ASYNC(n) __builtin_amdgcn_s_wait_asynccnt(n)
#else
#define WAIT_ASYNC(n) asm volatile("s_wait_asynccnt %0" ::"i"(n) : "memory")
#endif
#endif

// windowed token index r -> linear pixel row (b*4096 + H*64 + W)
__device__ __forceinline__ int win_to_linear(int r) {
    int win = r >> 6, t = r & 63;
    int b  = win >> 6;
    int wh = (win >> 3) & 7;
    int ww = win & 7;
    int i  = t >> 3;
    int j  = t & 7;
    return (b * 64 + wh * 8 + i) * 64 + (ww * 8 + j);
}

// ---------------------------------------------------------------------------
// Prep: fp32 -> f16 (vectorized), and fp32 [K][N] -> f16 [N][K] transpose.
// ---------------------------------------------------------------------------
__global__ __launch_bounds__(256) void cvt_f16(const float* __restrict__ in,
                                               _Float16* __restrict__ out, int n4) {
    int i = blockIdx.x * 256 + threadIdx.x;
    if (i < n4) {
        v4f f = ((const v4f*)in)[i];
        ((v4h*)out)[i] = __builtin_convertvector(f, v4h);
    }
}

__global__ __launch_bounds__(256) void cvt_transpose(const float* __restrict__ w,
                                                     _Float16* __restrict__ wt,
                                                     int K, int N) {
    int i = blockIdx.x * 256 + threadIdx.x;
    if (i < K * N) {
        int k = i / N, n = i - k * N;
        wt[(size_t)n * K + k] = (_Float16)w[i];
    }
}

// ---------------------------------------------------------------------------
// WMMA GEMM, K = 512, workgroup tile 128x64, 8 wave32, wave tile 32x32
// (4 v_wmma_f32_16x16x32_f16 per K-step). LDS double-buffered.
//   WIN_A:   gather A rows through win_to_linear (qkv reads pixel-order x)
//   OUT_F16: f16 output, identity rows (qkv) ; else f32 output, win-scattered
//            rows (proj un-windows on store).
// ---------------------------------------------------------------------------
template <bool WIN_A, bool OUT_F16>
__global__ __launch_bounds__(256)
void gemm_wmma(const _Float16* __restrict__ A,     // [32768][512]
               const _Float16* __restrict__ wT,    // [N][512]
               const float*    __restrict__ bias,  // [N]
               void*           __restrict__ outp,  // [32768][N]
               int N)
{
    constexpr int K = 512;
    __shared__ _Float16 As[2][128][32];
    __shared__ _Float16 Bs[2][64][32];

    const int tid  = threadIdx.x;
    const int lane = tid & 31, wv = tid >> 5;
    const int l = lane & 15, half = lane >> 4;
    const int m0 = blockIdx.y * 128;
    const int n0 = blockIdx.x * 64;
    const int wr = (wv >> 1) * 32;   // wave row offset in tile
    const int wc = (wv & 1) * 32;    // wave col offset in tile

    // A stage: 128 rows x 32 f16, 2 threads/row, 2x16B each
    const int arow = tid >> 1;
    const int acol = (tid & 1) * 16;
    const int amrow = m0 + arow;
    const _Float16* aptr =
        A + (size_t)(WIN_A ? win_to_linear(amrow) : amrow) * K + acol;
    // B stage: 64 rows x 32 f16, 4 threads/row, 16B each
    const int brow = tid >> 2;
    const int bcol = (tid & 3) * 8;
    const _Float16* bptr = wT + (size_t)(n0 + brow) * K + bcol;

    auto stage = [&](int buf, int k0) {
#if HAVE_ASYNC_LDS
        ASYNC_COPY16(aptr + k0,     &As[buf][arow][acol]);
        ASYNC_COPY16(aptr + k0 + 8, &As[buf][arow][acol + 8]);
        ASYNC_COPY16(bptr + k0,     &Bs[buf][brow][bcol]);
#else
        *(v8h*)&As[buf][arow][acol]     = *(const v8h*)(aptr + k0);
        *(v8h*)&As[buf][arow][acol + 8] = *(const v8h*)(aptr + k0 + 8);
        *(v8h*)&Bs[buf][brow][bcol]     = *(const v8h*)(bptr + k0);
#endif
    };

    v8f acc[2][2] = {};

    stage(0, 0);
    int cur = 0;
    for (int k0 = 0; k0 < K; k0 += 32) {
        const bool more = (k0 + 32) < K;
        if (more) stage(cur ^ 1, k0 + 32);          // overlap DMA with WMMA
        if (k0 + 64 < K) {                          // near-cache prefetch
            __builtin_prefetch(aptr + k0 + 64, 0, 3);
            __builtin_prefetch(bptr + k0 + 64, 0, 3);
        }
#if HAVE_ASYNC_LDS
        // async loads complete in order: <=3 outstanding => current buf ready
        if (more) WAIT_ASYNC(3); else WAIT_ASYNC(0);
#endif
        __syncthreads();

        V16 a[2], b[2];
#pragma unroll
        for (int i = 0; i < 2; ++i) {               // ISA 16-bit A layout
            int row = wr + 16 * i + l;
            a[i].h[0] = *(const v8h*)&As[cur][row][8 * half];
            a[i].h[1] = *(const v8h*)&As[cur][row][16 + 8 * half];
        }
#pragma unroll
        for (int j = 0; j < 2; ++j) {               // B via BT: contiguous K run
            int col = wc + 16 * j + l;
            b[j].h[0] = *(const v8h*)&Bs[cur][col][8 * half];
            b[j].h[1] = *(const v8h*)&Bs[cur][col][16 + 8 * half];
        }
#pragma unroll
        for (int i = 0; i < 2; ++i)
#pragma unroll
            for (int j = 0; j < 2; ++j)
                acc[i][j] = __builtin_amdgcn_wmma_f32_16x16x32_f16(
                    false, a[i].v, false, b[j].v, (short)0, acc[i][j], false, false);
        __syncthreads();                            // release buf before rewrite
        cur ^= 1;
    }

#pragma unroll
    for (int i = 0; i < 2; ++i) {
#pragma unroll
        for (int r = 0; r < 8; ++r) {
            int m = m0 + wr + 16 * i + 8 * half + r;   // C layout: M = r + 8*half
            if constexpr (OUT_F16) {
                _Float16* out = (_Float16*)outp;
#pragma unroll
                for (int j = 0; j < 2; ++j) {
                    int col = n0 + wc + 16 * j + l;    // N = lane%16
                    out[(size_t)m * N + col] = (_Float16)(acc[i][j][r] + bias[col]);
                }
            } else {
                float* out = (float*)outp;
                size_t lin = (size_t)win_to_linear(m) * N;  // un-window on store
#pragma unroll
                for (int j = 0; j < 2; ++j) {
                    int col = n0 + wc + 16 * j + l;
                    out[lin + col] = acc[i][j][r] + bias[col];
                }
            }
        }
    }
}

// ---------------------------------------------------------------------------
// Windowed attention: one block per (window, head). 4 wave32.
// q,k get axial 2D RoPE (h-pos on dims 0..15, w-pos on dims 16..31);
// q also pre-scaled by HD^-0.5. S = qk^T (wmma), softmax f32, O = P v (wmma).
// ---------------------------------------------------------------------------
__global__ __launch_bounds__(128)
void attn_win(const _Float16* __restrict__ qkvh,  // [32768][1536] windowed rows
              _Float16*       __restrict__ aout)  // [32768][512]  windowed rows
{
    __shared__ _Float16 Qs[64][32];
    __shared__ _Float16 Ks[64][32];
    __shared__ _Float16 Vt[32][64];   // v transposed: [hd][token]
    __shared__ float    Sc[64][64];
    __shared__ _Float16 Ps[64][64];

    const int bh  = blockIdx.x;
    const int b_  = bh >> 4;
    const int h   = bh & 15;
    const int tid = threadIdx.x;
    const int lane = tid & 31, wv = tid >> 5;
    const int l = lane & 15, half = lane >> 4;
    const float SCALE = 0.17677669529663687f;   // 32^-0.5

    // ---- stage q/k with RoPE, v transposed ----
    {
        const int n  = tid >> 1;        // token 0..63  (n = i*8 + j)
        const int ch = tid & 1;         // 0: height dims 0..15, 1: width dims 16..31
        const float pos = (float)(ch ? (n & 7) : (n >> 3));
        const size_t base = (size_t)(b_ * 64 + n) * 1536 + h * 32 + ch * 16;
        v8h q0 = *(const v8h*)(qkvh + base);
        v8h q1 = *(const v8h*)(qkvh + base + 8);
        v8h k0 = *(const v8h*)(qkvh + base + 512);
        v8h k1 = *(const v8h*)(qkvh + base + 520);
        v8h v0 = *(const v8h*)(qkvh + base + 1024);
        v8h v1 = *(const v8h*)(qkvh + base + 1032);
        v8h qo0, qo1, ko0, ko1;
#pragma unroll
        for (int c = 0; c < 8; ++c) {
            // inv_freq = 10000^(-2c/16); rotate pair (c, c+8)
            float inv = __powf(10000.0f, -(float)(2 * c) * 0.0625f);
            float ang = pos * inv;
            float cs = __cosf(ang), sn = __sinf(ang);
            float qa = (float)q0[c], qb = (float)q1[c];
            qo0[c] = (_Float16)((qa * cs - qb * sn) * SCALE);
            qo1[c] = (_Float16)((qb * cs + qa * sn) * SCALE);
            float ka = (float)k0[c], kb = (float)k1[c];
            ko0[c] = (_Float16)(ka * cs - kb * sn);
            ko1[c] = (_Float16)(kb * cs + ka * sn);
        }
        *(v8h*)&Qs[n][ch * 16]     = qo0;
        *(v8h*)&Qs[n][ch * 16 + 8] = qo1;
        *(v8h*)&Ks[n][ch * 16]     = ko0;
        *(v8h*)&Ks[n][ch * 16 + 8] = ko1;
#pragma unroll
        for (int c = 0; c < 8; ++c) {
            Vt[ch * 16 + c][n]     = v0[c];
            Vt[ch * 16 + c + 8][n] = v1[c];
        }
    }
    __syncthreads();

    // ---- S = q k^T : wave wv owns rows 16*wv..+15, 4 col tiles, K=32 ----
    {
        V16 a;
        const int row = 16 * wv + l;
        a.h[0] = *(const v8h*)&Qs[row][8 * half];
        a.h[1] = *(const v8h*)&Qs[row][16 + 8 * half];
#pragma unroll
        for (int ct = 0; ct < 4; ++ct) {
            V16 b;
            const int col = 16 * ct + l;          // BT = k's natural [token][hd]
            b.h[0] = *(const v8h*)&Ks[col][8 * half];
            b.h[1] = *(const v8h*)&Ks[col][16 + 8 * half];
            v8f s = {};
            s = __builtin_amdgcn_wmma_f32_16x16x32_f16(
                false, a.v, false, b.v, (short)0, s, false, false);
#pragma unroll
            for (int r = 0; r < 8; ++r)
                Sc[16 * wv + 8 * half + r][16 * ct + l] = s[r];
        }
    }
    __syncthreads();

    // ---- row softmax (f32), emit P in f16 ----
    if (tid < 64) {
        float m = -1e30f;
        for (int c = 0; c < 64; ++c) m = fmaxf(m, Sc[tid][c]);
        float s = 0.0f;
        for (int c = 0; c < 64; ++c) {
            float e = __expf(Sc[tid][c] - m);
            Sc[tid][c] = e;
            s += e;
        }
        float rs = 1.0f / s;
        for (int c = 0; c < 64; ++c) Ps[tid][c] = (_Float16)(Sc[tid][c] * rs);
    }
    __syncthreads();

    // ---- O = P v : rows 16*wv, N=32 (2 col tiles), K=64 (2 steps) ----
    v8f o[2] = {};
#pragma unroll
    for (int ks = 0; ks < 64; ks += 32) {
        V16 a;
        const int row = 16 * wv + l;
        a.h[0] = *(const v8h*)&Ps[row][ks + 8 * half];
        a.h[1] = *(const v8h*)&Ps[row][ks + 16 + 8 * half];
#pragma unroll
        for (int ct = 0; ct < 2; ++ct) {
            V16 b;
            const int nn = 16 * ct + l;           // BT = [hd][token] = Vt
            b.h[0] = *(const v8h*)&Vt[nn][ks + 8 * half];
            b.h[1] = *(const v8h*)&Vt[nn][ks + 16 + 8 * half];
            o[ct] = __builtin_amdgcn_wmma_f32_16x16x32_f16(
                false, a.v, false, b.v, (short)0, o[ct], false, false);
        }
    }
#pragma unroll
    for (int ct = 0; ct < 2; ++ct)
#pragma unroll
        for (int r = 0; r < 8; ++r)
            aout[(size_t)(b_ * 64 + 16 * wv + 8 * half + r) * 512 + h * 32 + 16 * ct + l]
                = (_Float16)o[ct][r];
}

// ---------------------------------------------------------------------------
// Host launcher
// ---------------------------------------------------------------------------
extern "C" void kernel_launch(void* const* d_in, const int* in_sizes, int n_in,
                              void* d_out, int out_size, void* d_ws, size_t ws_size,
                              hipStream_t stream)
{
    (void)in_sizes; (void)n_in; (void)out_size; (void)ws_size;
    const float* x      = (const float*)d_in[0];
    const float* qkv_w  = (const float*)d_in[1];
    const float* qkv_b  = (const float*)d_in[2];
    // d_in[3] kv_w, d_in[4] kv_b: hy_kv branch is unused by the reference output.
    const float* proj_w = (const float*)d_in[5];
    const float* proj_b = (const float*)d_in[6];
    float* out = (float*)d_out;

    char* ws = (char*)d_ws;
    _Float16* xh     = (_Float16*)(ws);                 // 32768x512  f16 (32 MB)
    _Float16* qkvwT  = (_Float16*)(ws + 33554432ull);   // 1536x512   f16
    _Float16* projwT = (_Float16*)(ws + 35127296ull);   // 512x512    f16
    _Float16* qkvh   = (_Float16*)(ws + 35651584ull);   // 32768x1536 f16 (96 MB)
    _Float16* attnh  = (_Float16*)(ws + 136314880ull);  // 32768x512  f16 (32 MB)

    cvt_f16<<<16384, 256, 0, stream>>>(x, xh, 4194304);                 // 16.7M / 4
    cvt_transpose<<<3072, 256, 0, stream>>>(qkv_w, qkvwT, 512, 1536);
    cvt_transpose<<<1024, 256, 0, stream>>>(proj_w, projwT, 512, 512);

    gemm_wmma<true,  true ><<<dim3(24, 256), 256, 0, stream>>>(xh, qkvwT, qkv_b, qkvh, 1536);
    attn_win<<<8192, 128, 0, stream>>>(qkvh, attnh);                    // 512 win x 16 heads
    gemm_wmma<false, false><<<dim3(8, 256), 256, 0, stream>>>(attnh, projwT, proj_b, out, 512);
}
